// ReLUFFN_32409823216253
// MI455X (gfx1250) — 8-bit matrix operand order repaired and checked by reading, compiled, not run on hardware
//
#include <hip/hip_runtime.h>
#include <stdint.h>

typedef int v8i __attribute__((ext_vector_type(8)));

#define EPS 1e-5f

__global__ __launch_bounds__(256) void absmean_partial(
    const float* __restrict__ W, size_t n, float* __restrict__ partial) {
  float s = 0.0f;
  for (size_t i = (size_t)blockIdx.x * blockDim.x + threadIdx.x; i < n;
       i += (size_t)gridDim.x * blockDim.x)
    s += fabsf(W[i]);
  __shared__ float red[256];
  red[threadIdx.x] = s;
  __syncthreads();
  for (int st = 128; st > 0; st >>= 1) {
    if ((int)threadIdx.x < st) red[threadIdx.x] += red[threadIdx.x + st];
    __syncthreads();
  }
  if (threadIdx.x == 0) partial[blockIdx.x] = red[0];
}

__global__ __launch_bounds__(256) void absmean_final(
    const float* __restrict__ part, float* __restrict__ scales, float inv_n) {
  __shared__ float red[256];
  for (int j = 0; j < 3; ++j) {
    float s = 0.0f;
    for (int i = threadIdx.x; i < 1024; i += 256) s += part[j * 1024 + i];
    red[threadIdx.x] = s;
    __syncthreads();
    for (int st = 128; st > 0; st >>= 1) {
      if ((int)threadIdx.x < st) red[threadIdx.x] += red[threadIdx.x + st];
      __syncthreads();
    }
    if (threadIdx.x == 0) {
      float mean = red[0] * inv_n;
      float d = fmaxf(mean, EPS);
      scales[j] = 1.0f / d;
      scales[3 + j] = d;
    }
    __syncthreads();
  }
}

__global__ __launch_bounds__(256) void quant_pack_A(
    const float* __restrict__ X, int D, int8_t* __restrict__ Q,
    float* __restrict__ dscale) {
  const int t = blockIdx.x;
  const int tid = threadIdx.x;
  const float* row = X + (size_t)t * D;

  float amax = 0.0f;
  for (int k = tid; k < D; k += 256) amax = fmaxf(amax, fabsf(row[k]));
  __shared__ float red[256];
  red[tid] = amax;
  __syncthreads();
  for (int st = 128; st > 0; st >>= 1) {
    if (tid < st) red[tid] = fmaxf(red[tid], red[tid + st]);
    __syncthreads();
  }
  const float mx = fmaxf(red[0], EPS);
  const float scale = 127.0f / mx;
  if (tid == 0) dscale[t] = mx * (1.0f / 127.0f);

  const int KT = D >> 6;
  const int mt = t >> 4, m = t & 15;
  uint32_t* Qd = (uint32_t*)Q;
  for (int g = tid; g < (D >> 2); g += 256) {
    const int k = g << 2;
    const float4 xv = *(const float4*)(row + k);
    int q0 = max(-128, min(127, (int)rintf(xv.x * scale)));
    int q1 = max(-128, min(127, (int)rintf(xv.y * scale)));
    int q2 = max(-128, min(127, (int)rintf(xv.z * scale)));
    int q3 = max(-128, min(127, (int)rintf(xv.w * scale)));
    uint32_t dw = (uint32_t)(q0 & 255) | ((uint32_t)(q1 & 255) << 8) |
                  ((uint32_t)(q2 & 255) << 16) | ((uint32_t)(q3 & 255) << 24);
    const int kt = k >> 6, kq = k & 63;
    const int half = (kq >> 3) & 1;
    const int v = ((kq >> 4) << 1) | ((kq & 7) >> 2);
    const int lanei = m | (half << 4);
    Qd[(((size_t)mt * KT + kt) << 8) + (lanei << 3) + v] = dw;
  }
}

__global__ __launch_bounds__(256) void quant_pack_B(
    const float* __restrict__ W, int N, int K,
    const float* __restrict__ scales, int sidx, int8_t* __restrict__ Q) {
  const float s = scales[sidx];
  const size_t total = ((size_t)N * K) >> 2;
  const int KG = K >> 2, KT = K >> 6;
  uint32_t* Qd = (uint32_t*)Q;
  for (size_t i = (size_t)blockIdx.x * blockDim.x + threadIdx.x; i < total;
       i += (size_t)gridDim.x * blockDim.x) {
    const int kg = (int)(i % KG);
    const int n = (int)(i / KG);
    const int k = kg << 2;
    const float4 wv = *(const float4*)(W + (size_t)n * K + k);
    int q0 = max(-1, min(1, (int)rintf(wv.x * s)));
    int q1 = max(-1, min(1, (int)rintf(wv.y * s)));
    int q2 = max(-1, min(1, (int)rintf(wv.z * s)));
    int q3 = max(-1, min(1, (int)rintf(wv.w * s)));
    uint32_t dw = (uint32_t)(q0 & 255) | ((uint32_t)(q1 & 255) << 8) |
                  ((uint32_t)(q2 & 255) << 16) | ((uint32_t)(q3 & 255) << 24);
    const int kt = k >> 6, kq = k & 63;
    const int khalf = (kq >> 3) & 1;
    const int v = ((kq >> 4) << 1) | ((kq & 7) >> 2);
    const int lanei = (n & 15) | (khalf << 4);
    Qd[(((size_t)(n >> 4) * KT + kt) << 8) + (lanei << 3) + v] = dw;
  }
}

__global__ __launch_bounds__(256) void gemm_gateup(
    const int8_t* __restrict__ Aq, const int8_t* __restrict__ Bg,
    const int8_t* __restrict__ Bu, const float* __restrict__ ads,
    const float* __restrict__ scales, float* __restrict__ H, int F, int K) {
  const int KT = K >> 6;
  const int lane = threadIdx.x & 31;
  const int wv = threadIdx.x >> 5;
  const int wm = wv >> 2, wn = wv & 3;
  const int t0 = blockIdx.x << 6;
  const int f0 = blockIdx.y << 7;
  const int tw = t0 + (wm << 5);
  const int fw = f0 + (wn << 5);

  __shared__ float adsS[64];
  if (threadIdx.x < 64) adsS[threadIdx.x] = ads[t0 + threadIdx.x];
  __syncthreads();

  const float gds = scales[3], uds = scales[4];

  const v8i* pa0 = (const v8i*)Aq + ((size_t)((tw >> 4) + 0) * KT) * 32 + lane;
  const v8i* pa1 = (const v8i*)Aq + ((size_t)((tw >> 4) + 1) * KT) * 32 + lane;
  const v8i* pg0 = (const v8i*)Bg + ((size_t)((fw >> 4) + 0) * KT) * 32 + lane;
  const v8i* pg1 = (const v8i*)Bg + ((size_t)((fw >> 4) + 1) * KT) * 32 + lane;
  const v8i* pu0 = (const v8i*)Bu + ((size_t)((fw >> 4) + 0) * KT) * 32 + lane;
  const v8i* pu1 = (const v8i*)Bu + ((size_t)((fw >> 4) + 1) * KT) * 32 + lane;

  v8i z = {0, 0, 0, 0, 0, 0, 0, 0};
  v8i accg00 = z, accg01 = z, accg10 = z, accg11 = z;
  v8i accu00 = z, accu01 = z, accu10 = z, accu11 = z;

#pragma unroll 2
  for (int kt = 0; kt < KT; ++kt) {
    const size_t ko = (size_t)kt * 32;
    v8i a0 = pa0[ko];
    v8i a1 = pa1[ko];
    v8i g0 = pg0[ko];
    v8i g1 = pg1[ko];
    v8i u0 = pu0[ko];
    v8i u1 = pu1[ko];
    __builtin_prefetch((const void*)(pg0 + ko + 64), 0, 3);
    __builtin_prefetch((const void*)(pu0 + ko + 64), 0, 3);
    accg00 = __builtin_amdgcn_wmma_i32_16x16x64_iu8(true, a0, true, g0, accg00, false, false);
    accg01 = __builtin_amdgcn_wmma_i32_16x16x64_iu8(true, a0, true, g1, accg01, false, false);
    accg10 = __builtin_amdgcn_wmma_i32_16x16x64_iu8(true, a1, true, g0, accg10, false, false);
    accg11 = __builtin_amdgcn_wmma_i32_16x16x64_iu8(true, a1, true, g1, accg11, false, false);
    accu00 = __builtin_amdgcn_wmma_i32_16x16x64_iu8(true, a0, true, u0, accu00, false, false);
    accu01 = __builtin_amdgcn_wmma_i32_16x16x64_iu8(true, a0, true, u1, accu01, false, false);
    accu10 = __builtin_amdgcn_wmma_i32_16x16x64_iu8(true, a1, true, u0, accu10, false, false);
    accu11 = __builtin_amdgcn_wmma_i32_16x16x64_iu8(true, a1, true, u1, accu11, false, false);
  }

  const int half = lane >> 4, col = lane & 15;
  const v8i* AG[2][2] = {{&accg00, &accg01}, {&accg10, &accg11}};
  const v8i* AU[2][2] = {{&accu00, &accu01}, {&accu10, &accu11}};
#pragma unroll
  for (int mi = 0; mi < 2; ++mi)
#pragma unroll
    for (int ni = 0; ni < 2; ++ni) {
      const int fcol = fw + (ni << 4) + col;
#pragma unroll
      for (int r = 0; r < 8; ++r) {
        const int trow = tw + (mi << 4) + r + (half << 3);
        const float asc = adsS[trow - t0];
        float g = (float)((*AG[mi][ni])[r]) * asc * gds;
        g = fmaxf(g, 0.0f);
        const float u = (float)((*AU[mi][ni])[r]) * asc * uds;
        H[(size_t)trow * F + fcol] = g * u;
      }
    }
}

__global__ __launch_bounds__(256) void gemm_down(
    const int8_t* __restrict__ Aq, const int8_t* __restrict__ Bd,
    const float* __restrict__ ads, const float* __restrict__ scales,
    float* __restrict__ Out, int N, int K) {
  const int KT = K >> 6;
  const int lane = threadIdx.x & 31;
  const int wv = threadIdx.x >> 5;
  const int wm = wv >> 2, wn = wv & 3;
  const int t0 = blockIdx.x << 6;
  const int n0 = blockIdx.y << 7;
  const int tw = t0 + (wm << 5);
  const int nw = n0 + (wn << 5);

  __shared__ float adsS[64];
  if (threadIdx.x < 64) adsS[threadIdx.x] = ads[t0 + threadIdx.x];
  __syncthreads();

  const float dds = scales[5];

  const v8i* pa0 = (const v8i*)Aq + ((size_t)((tw >> 4) + 0) * KT) * 32 + lane;
  const v8i* pa1 = (const v8i*)Aq + ((size_t)((tw >> 4) + 1) * KT) * 32 + lane;
  const v8i* pb0 = (const v8i*)Bd + ((size_t)((nw >> 4) + 0) * KT) * 32 + lane;
  const v8i* pb1 = (const v8i*)Bd + ((size_t)((nw >> 4) + 1) * KT) * 32 + lane;

  v8i z = {0, 0, 0, 0, 0, 0, 0, 0};
  v8i acc00 = z, acc01 = z, acc10 = z, acc11 = z;

#pragma unroll 2
  for (int kt = 0; kt < KT; ++kt) {
    const size_t ko = (size_t)kt * 32;
    v8i a0 = pa0[ko];
    v8i a1 = pa1[ko];
    v8i b0 = pb0[ko];
    v8i b1 = pb1[ko];
    __builtin_prefetch((const void*)(pa0 + ko + 64), 0, 3);
    __builtin_prefetch((const void*)(pb0 + ko + 64), 0, 3);
    acc00 = __builtin_amdgcn_wmma_i32_16x16x64_iu8(true, a0, true, b0, acc00, false, false);
    acc01 = __builtin_amdgcn_wmma_i32_16x16x64_iu8(true, a0, true, b1, acc01, false, false);
    acc10 = __builtin_amdgcn_wmma_i32_16x16x64_iu8(true, a1, true, b0, acc10, false, false);
    acc11 = __builtin_amdgcn_wmma_i32_16x16x64_iu8(true, a1, true, b1, acc11, false, false);
  }

  const int half = lane >> 4, col = lane & 15;
  const v8i* AC[2][2] = {{&acc00, &acc01}, {&acc10, &acc11}};
#pragma unroll
  for (int mi = 0; mi < 2; ++mi)
#pragma unroll
    for (int ni = 0; ni < 2; ++ni) {
      const int ncol = nw + (ni << 4) + col;
#pragma unroll
      for (int r = 0; r < 8; ++r) {
        const int trow = tw + (mi << 4) + r + (half << 3);
        const float asc = adsS[trow - t0];
        Out[(size_t)trow * N + ncol] = (float)((*AC[mi][ni])[r]) * asc * dds;
      }
    }
}

extern "C" void kernel_launch(void* const* d_in, const int* in_sizes, int n_in,
                              void* d_out, int out_size, void* d_ws,
                              size_t ws_size, hipStream_t stream) {
  (void)in_sizes; (void)n_in; (void)out_size; (void)ws_size;
  const float* x = (const float*)d_in[0];
  const float* w_gate = (const float*)d_in[1];
  const float* w_up = (const float*)d_in[2];
  const float* w_down = (const float*)d_in[3];
  float* out = (float*)d_out;

  constexpr int Bb = 4, S = 2048, D = 2048, F = 8192;
  constexpr int T = Bb * S;
  constexpr size_t nW = (size_t)F * D;

  uint8_t* ws = (uint8_t*)d_ws;
  auto alloc = [&](size_t bytes) -> uint8_t* {
    uint8_t* p = ws;
    ws += (bytes + 255) & ~(size_t)255;
    return p;
  };
  int8_t* xq = (int8_t*)alloc((size_t)T * D);
  int8_t* wgq = (int8_t*)alloc(nW);
  int8_t* wuq = (int8_t*)alloc(nW);
  int8_t* wdq = (int8_t*)alloc(nW);
  float* h = (float*)alloc((size_t)T * F * 4);
  int8_t* hq = (int8_t*)alloc((size_t)T * F);
  float* ads = (float*)alloc((size_t)T * 4);
  float* hds = (float*)alloc((size_t)T * 4);
  float* part = (float*)alloc(3 * 1024 * 4);
  float* scales = (float*)alloc(6 * 4);

  absmean_partial<<<1024, 256, 0, stream>>>(w_gate, nW, part);
  absmean_partial<<<1024, 256, 0, stream>>>(w_up, nW, part + 1024);
  absmean_partial<<<1024, 256, 0, stream>>>(w_down, nW, part + 2048);
  absmean_final<<<1, 256, 0, stream>>>(part, scales, 1.0f / (float)nW);

  quant_pack_A<<<T, 256, 0, stream>>>(x, D, xq, ads);
  quant_pack_B<<<2048, 256, 0, stream>>>(w_gate, F, D, scales, 0, wgq);
  quant_pack_B<<<2048, 256, 0, stream>>>(w_up, F, D, scales, 1, wuq);
  quant_pack_B<<<2048, 256, 0, stream>>>(w_down, D, F, scales, 2, wdq);

  gemm_gateup<<<dim3(T / 64, F / 128), 256, 0, stream>>>(xq, wgq, wuq, ads,
                                                         scales, h, F, D);

  quant_pack_A<<<T, 256, 0, stream>>>(h, F, hq, hds);

  gemm_down<<<dim3(T / 64, D / 128), 256, 0, stream>>>(hq, wdq, hds, scales,
                                                       out, D, F);
}
